// DistortionConvLayer_81355270521450
// MI455X (gfx1250) — compile-verified, matrix operand order checked by reading, not run on hardware
//
#include <hip/hip_runtime.h>

// ---------- problem constants ----------
#define Bq   4
#define Hh   128
#define Ww   256
#define Cc   64
#define Ff   128
#define KTOT 576          // 9 taps * 64 channels
#define LROW 292          // dwords per LDS A row (584 bf16): rows stay 16B aligned
#define LDS_BYTES (64 * LROW * 4)   // 74752 B dynamic LDS (4 blocks/WGP at 320KB)

typedef __attribute__((ext_vector_type(16))) __bf16 v16bf;
typedef __attribute__((ext_vector_type(8)))  float  v8f;

union ABfrag { uint4 u[2]; v16bf v; };

static __device__ __forceinline__ unsigned int pack_bf16x2(float lo, float hi) {
    unsigned int ul = __float_as_uint(lo);
    unsigned int uh = __float_as_uint(hi);
    ul = (ul + 0x7FFFu + ((ul >> 16) & 1u)) >> 16;   // RNE f32->bf16
    uh = (uh + 0x7FFFu + ((uh >> 16) & 1u)) >> 16;
    return (ul & 0xFFFFu) | (uh << 16);
}

// ---------- prep 1: per-row distortion offsets (faithful port of make_offset) ----------
__global__ void dcl_offsets_kernel(float2* __restrict__ offs /* [H*9] */) {
    int h = blockIdx.x * blockDim.x + threadIdx.x;
    if (h >= Hh) return;
    const float PI = 3.14159265358979323846f;
    const float unit_w = 2.0f * PI / (float)Ww;
    const float unit_h = PI / (2.0f * (float)Hh);      // skydome
    const float rho = tanf(unit_w);
    const float theta = ((float)(Ww / 2) - 0.5f * (float)Ww) * unit_w;  // == 0
    const float phi = (float)(Hh - h) * unit_h;
    float cphi = cosf(phi), sphi = sinf(phi);
    float px = cphi * cosf(theta), py = sphi, pz = cphi * sinf(theta);
    // t_x = v x p_u, v = (0,1,0)
    float txx = pz, txy = 0.0f, txz = -px;
    // t_y = p_u x t_x
    float tyx = py * txz - pz * txy;
    float tyy = pz * txx - px * txz;
    float tyz = px * txy - py * txx;
    const int rg0[9] = { 1, 1, 1, 0, 0, 0, -1, -1, -1 };
    const int rg1[9] = { -1, 0, 1, -1, 0, 1, -1, 0, 1 };
    float xr[9], yr[9];
#pragma unroll
    for (int t = 0; t < 9; ++t) {
        float r0 = (float)rg0[t], r1 = (float)rg1[t];
        float ux = px + rho * (r0 * txx + r1 * tyx);
        float uy = py + rho * (r0 * txy + r1 * tyy);
        float uz = pz + rho * (r0 * txz + r1 * tyz);
        float base = atan2f(uz, ux);
        float tr;
        if (ux > 0.0f)       tr = base;
        else if (ux < 0.0f)  tr = (uz >= 0.0f) ? base + PI : base - PI;
        else                 tr = (uz > 0.0f) ? 0.5f * PI : -0.5f * PI;
        float phr = asinf(uy);
        xr[t] = (tr / PI + 1.0f) * 0.5f * (float)Ww;
        yr[t] = (1.0f - 2.0f * phr / PI) * (float)Hh;
    }
#pragma unroll
    for (int t = 0; t < 9; ++t) {
        // faithful to reference: component 0 (x_r diff) is later added to y, comp 1 to x
        offs[h * 9 + t] = make_float2(xr[t] - xr[4], yr[t] - yr[4]);
    }
}

// ---------- prep 2: weights f32[576][128] -> bf16 Bt[n][k] (transposed) ----------
__global__ void dcl_wprep_kernel(const float* __restrict__ kern,
                                 unsigned short* __restrict__ Bt) {
    int i = blockIdx.x * 256 + threadIdx.x;
    if (i >= KTOT * Ff) return;
    int k = i / Ff, n = i - k * Ff;
    unsigned int u = __float_as_uint(kern[i]);
    u = (u + 0x7FFFu + ((u >> 16) & 1u)) >> 16;
    Bt[(size_t)n * KTOT + k] = (unsigned short)u;
}

// ---------- main: gather+bilinear -> LDS bf16 A-tile -> WMMA GEMM ----------
__global__ __launch_bounds__(256, 1) void dcl_main_kernel(
    const float* __restrict__ in, const float2* __restrict__ offs,
    const unsigned short* __restrict__ Bt, const float* __restrict__ bias,
    float* __restrict__ out) {
    extern __shared__ unsigned int lds[];   // A-tile: [64 pixels][LROW dwords]

    const int bid  = blockIdx.x;            // 2048 blocks: b*512 + y*4 + xblk
    const int xblk = bid & 3;
    const int y    = (bid >> 2) & (Hh - 1);
    const int b    = bid >> 9;
    const int tid  = threadIdx.x;

    // ---- phase 1: build bf16 A-tile [64 px][576 K] via bilinear gather ----
    const int in_h = Hh + 2, in_w = Ww + 2;
    for (int pt = tid; pt < 64 * 9; pt += 256) {
        int p = pt / 9, t = pt - p * 9;
        int gx = xblk * 64 + p;
        float2 o = offs[y * 9 + t];
        float ys = (float)y  + (float)(t / 3) + o.x;   // x_r diff added to y (faithful)
        float xs = (float)gx + (float)(t % 3) + o.y;   // y_r diff added to x (faithful)
        ys = fminf(fmaxf(ys, 0.0f), (float)(in_h - 1));
        if (xs < 0.0f)              xs += (float)in_w;
        if (xs > (float)(in_w - 1)) xs -= (float)in_w;
        int y0 = (int)floorf(ys), x0 = (int)floorf(xs);
        int y1 = y0 + 1, x1 = x0 + 1;
        y0 = min(max(y0, 0), in_h - 1); y1 = min(max(y1, 0), in_h - 1);
        x0 = min(max(x0, 0), in_w - 1); x1 = min(max(x1, 0), in_w - 1);
        float y0f = (float)y0, y1f = (float)y1, x0f = (float)x0, x1f = (float)x1;
        float w0 = (y1f - ys) * (x1f - xs);
        float w1 = (y1f - ys) * (xs - x0f);
        float w2 = (ys - y0f) * (x1f - xs);
        float w3 = (ys - y0f) * (xs - x0f);
        // padded-space coord -> input pointer (zero pad outside [1..H]x[1..W])
        const float4* q0 = (y0 >= 1 && y0 <= Hh && x0 >= 1 && x0 <= Ww)
            ? (const float4*)(in + (((size_t)b * Hh + (y0 - 1)) * Ww + (x0 - 1)) * Cc) : nullptr;
        const float4* q1 = (y0 >= 1 && y0 <= Hh && x1 >= 1 && x1 <= Ww)
            ? (const float4*)(in + (((size_t)b * Hh + (y0 - 1)) * Ww + (x1 - 1)) * Cc) : nullptr;
        const float4* q2 = (y1 >= 1 && y1 <= Hh && x0 >= 1 && x0 <= Ww)
            ? (const float4*)(in + (((size_t)b * Hh + (y1 - 1)) * Ww + (x0 - 1)) * Cc) : nullptr;
        const float4* q3 = (y1 >= 1 && y1 <= Hh && x1 >= 1 && x1 <= Ww)
            ? (const float4*)(in + (((size_t)b * Hh + (y1 - 1)) * Ww + (x1 - 1)) * Cc) : nullptr;
        const float4 z4 = make_float4(0.f, 0.f, 0.f, 0.f);
        unsigned int* dst = lds + (size_t)p * LROW + (t * Cc) / 2;
#pragma unroll 4
        for (int c4 = 0; c4 < Cc / 4; ++c4) {
            float4 a = q0 ? q0[c4] : z4;
            float4 bq = q1 ? q1[c4] : z4;
            float4 c = q2 ? q2[c4] : z4;
            float4 d = q3 ? q3[c4] : z4;
            float r0 = w0 * a.x + w1 * bq.x + w2 * c.x + w3 * d.x;
            float r1 = w0 * a.y + w1 * bq.y + w2 * c.y + w3 * d.y;
            float r2 = w0 * a.z + w1 * bq.z + w2 * c.z + w3 * d.z;
            float r3 = w0 * a.w + w1 * bq.w + w2 * c.w + w3 * d.w;
            dst[c4 * 2 + 0] = pack_bf16x2(r0, r1);
            dst[c4 * 2 + 1] = pack_bf16x2(r2, r3);
        }
    }
    __syncthreads();

    // ---- phase 2: WMMA GEMM, double-buffered software pipeline ----
    // 8 waves = 4(M-tiles of 16 px) x 2(N-halves of 64). Each wave: 16x64 tile.
    const int lane = tid & 31;
    const int wv   = tid >> 5;
    const int mt   = wv & 3;          // M tile index
    const int nh   = wv >> 2;         // N half index
    const int p0   = mt * 16;
    const int nbase = nh * 64;
    const int lr   = lane & 15;
    const int half = lane >> 4;

    v8f acc0 = {}, acc1 = {}, acc2 = {}, acc3 = {};
    const unsigned int* abase = lds + (size_t)(p0 + lr) * LROW + half * 4;
    const unsigned short* brow[4];
#pragma unroll
    for (int nt = 0; nt < 4; ++nt)
        brow[nt] = Bt + (size_t)(nbase + nt * 16 + lr) * KTOT + half * 8;

    // double buffers: A frag from LDS (2x8 VGPRs), 4 B frags from global (2x32 VGPRs)
    ABfrag af[2];
    ABfrag bfr[2][4];

    auto LOAD_A = [&](int buf, int kc) {
        af[buf].u[0] = *(const uint4*)(abase + kc / 2);       // K = kc + half*8 .. +7
        af[buf].u[1] = *(const uint4*)(abase + kc / 2 + 8);   // K = kc + 16 + half*8
    };
    auto LOAD_B = [&](int buf, int nt, int kc) {
        bfr[buf][nt].u[0] = *(const uint4*)(brow[nt] + kc);
        bfr[buf][nt].u[1] = *(const uint4*)(brow[nt] + kc + 16);
    };

    LOAD_A(0, 0);
#pragma unroll
    for (int nt = 0; nt < 4; ++nt) LOAD_B(0, nt, 0);

#pragma unroll
    for (int ks = 0; ks < KTOT / 32; ++ks) {
        const int cur = ks & 1, nxt = cur ^ 1;
        const int kn  = (ks + 1) * 32;
        const bool more = ks < KTOT / 32 - 1;
        // prefetch next step split around first WMMA to maximize load->use distance
        if (more) { LOAD_A(nxt, kn); LOAD_B(nxt, 0, kn); LOAD_B(nxt, 1, kn); }
        acc0 = __builtin_amdgcn_wmma_f32_16x16x32_bf16(false, af[cur].v, false,
                   bfr[cur][0].v, (short)0, acc0, false, false);
        if (more) { LOAD_B(nxt, 2, kn); LOAD_B(nxt, 3, kn); }
        acc1 = __builtin_amdgcn_wmma_f32_16x16x32_bf16(false, af[cur].v, false,
                   bfr[cur][1].v, (short)0, acc1, false, false);
        acc2 = __builtin_amdgcn_wmma_f32_16x16x32_bf16(false, af[cur].v, false,
                   bfr[cur][2].v, (short)0, acc2, false, false);
        acc3 = __builtin_amdgcn_wmma_f32_16x16x32_bf16(false, af[cur].v, false,
                   bfr[cur][3].v, (short)0, acc3, false, false);
    }

    // ---- epilogue: bias + ReLU, non-temporal f32 stores per C/D layout ----
    const int gx0 = xblk * 64 + p0 + half * 8;
    const size_t rb = (((size_t)b * Hh + y) * Ww + gx0) * Ff;
    v8f accs[4] = { acc0, acc1, acc2, acc3 };
#pragma unroll
    for (int nt = 0; nt < 4; ++nt) {
        int col = nbase + nt * 16 + lr;
        float bv = bias[col];
        v8f a = accs[nt];
#pragma unroll
        for (int v = 0; v < 8; ++v)
            __builtin_nontemporal_store(fmaxf(a[v] + bv, 0.0f),
                                        &out[rb + (size_t)v * Ff + col]);
    }
}

extern "C" void kernel_launch(void* const* d_in, const int* in_sizes, int n_in,
                              void* d_out, int out_size, void* d_ws, size_t ws_size,
                              hipStream_t stream) {
    const float* inputs = (const float*)d_in[0];
    const float* kern   = (const float*)d_in[1];
    const float* bias   = (const float*)d_in[2];
    float* out = (float*)d_out;

    float2* offs = (float2*)d_ws;                                   // 128*9*8 = 9216 B
    unsigned short* Bt = (unsigned short*)((char*)d_ws + 16384);    // 576*128*2 B

    dcl_offsets_kernel<<<1, 128, 0, stream>>>(offs);
    dcl_wprep_kernel<<<(KTOT * Ff + 255) / 256, 256, 0, stream>>>(kern, Bt);
    dcl_main_kernel<<<Bq * Hh * (Ww / 64), 256, LDS_BYTES, stream>>>(
        inputs, offs, Bt, bias, out);
}
